// Int8Linear_2997887172646
// MI455X (gfx1250) — matrix operand order repaired and checked by reading, not compiled, not run on hardware
//
#include <hip/hip_runtime.h>

// ---- problem constants (from reference) ----
#define IN_F   4096
#define OUT_F  11008
#define GS     128
#define NGRP   (IN_F / GS)      // 32

// ---- tiling ----
#define BM 64                   // block M tile
#define BN 128                  // block N tile
#define LDS_PITCH 136           // 128 halfs + 8 pad (bank-conflict stagger), 272B rows (16B aligned)

typedef __attribute__((ext_vector_type(16))) _Float16 v16h;
typedef __attribute__((ext_vector_type(8)))  _Float16 v8h;
typedef __attribute__((ext_vector_type(4)))  _Float16 v4h;
typedef __attribute__((ext_vector_type(8)))  float    v8f;
typedef __attribute__((ext_vector_type(4)))  int      v4i;
typedef __attribute__((ext_vector_type(2)))  float    v2f;

__global__ __launch_bounds__(256, 2)
void w8g128_linear_wmma(const float* __restrict__ x,
                        const int*   __restrict__ w,
                        const float* __restrict__ saz,   // [NGRP][OUT_F][2] (scale, zero)
                        const float* __restrict__ bias,  // [OUT_F]
                        float*       __restrict__ out)
{
    __shared__ __align__(16) _Float16 sA[BM * LDS_PITCH];

    const int tid   = threadIdx.x;
    const int wid   = tid >> 5;
    const int lane  = tid & 31;
    const int l16   = lane & 15;
    const bool hiHalf = (lane >= 16);

    const int blockM = blockIdx.y * BM;
    const int blockN = blockIdx.x * BN;
    const int waveM  = (wid & 1) * 32;        // 2 M-waves
    const int waveN  = (wid >> 1) * 32;       // 4 N-waves
    const int gm     = blockM + waveM;        // wave's global row base
    const int gn     = blockN + waveN;        // wave's global col base

    // output accumulators: [mi][nj], 2x2 fragments of 16x16
    v8f acc[2][2] = {};

    // all-ones B fragment (for per-group row sums in C layout)
    v16h bones;
#pragma unroll
    for (int i = 0; i < 16; ++i) bones[i] = (_Float16)1.0f;

    for (int g = 0; g < NGRP; ++g) {
        const int k0 = g * GS;

        // ---- cooperative stage: x[blockM..+64][k0..+128] f32 -> f16 in LDS ----
        __syncthreads();
        {
            const float4* xv = (const float4*)(x + (size_t)blockM * IN_F + k0);
            // 64 rows x 32 float4/row = 2048 float4, 256 threads -> 8 each
#pragma unroll
            for (int it = 0; it < 8; ++it) {
                const int idx = it * 256 + tid;
                const int row = idx >> 5;         // / 32 float4 per row
                const int c4  = idx & 31;
                const float4 v = xv[(size_t)row * (IN_F / 4) + c4];
                v4h hv = { (_Float16)v.x, (_Float16)v.y, (_Float16)v.z, (_Float16)v.w };
                *(v4h*)(&sA[row * LDS_PITCH + c4 * 4]) = hv;
            }
        }
        __syncthreads();

        // prefetch next group's weight rows for this wave's columns
        if (g + 1 < NGRP) {
            const char* pf = (const char*)(w + (size_t)(gn + l16) * IN_F + k0 + GS);
            __builtin_prefetch(pf, 0, 1);
        }

        // group-local accumulators
        v8f cg[2][2] = {};
        v8f csum[2]  = {};

#pragma unroll
        for (int ks = 0; ks < 4; ++ks) {        // 4 x K=32 WMMA steps per group
            // ---- A fragments from LDS (contiguous k = 16h + s, same order as B) ----
            // lane<16:  M=l16, K: [0..15]
            // lane>=16: M=l16, K: [16..31]
            const int khalf = hiHalf ? 16 : 0;
            v16h a[2];
#pragma unroll
            for (int mi = 0; mi < 2; ++mi) {
                const int lrow = waveM + mi * 16 + l16;
                union { v16h v; v8h h[2]; } u;
                u.h[0] = *(const v8h*)(&sA[lrow * LDS_PITCH + ks * 32 + khalf]);
                u.h[1] = *(const v8h*)(&sA[lrow * LDS_PITCH + ks * 32 + 8 + khalf]);
                a[mi] = u.v;
            }

            // ---- B fragments direct from global (raw int8 -> f16, exact) ----
            // lane<16:  N=l16, K = [0..15] ; lane>=16: N=l16, K = [16..31]
            // int values are in [-128,127]: route the convert through i16 so it
            // lowers to a single v_cvt_f16_i16 per element (exact), instead of
            // v_cvt_f32_i32 + v_cvt_pk_f16_f32.
            const int koff = hiHalf ? 16 : 0;
            v16h b[2];
#pragma unroll
            for (int nj = 0; nj < 2; ++nj) {
                const int n = gn + nj * 16 + l16;
                const v4i* wp = (const v4i*)(w + (size_t)n * IN_F + k0 + ks * 32 + koff);
                const v4i w0 = wp[0], w1 = wp[1], w2 = wp[2], w3 = wp[3];
                v16h bh;
                bh[0]  = (_Float16)(short)w0.x;  bh[1]  = (_Float16)(short)w0.y;
                bh[2]  = (_Float16)(short)w0.z;  bh[3]  = (_Float16)(short)w0.w;
                bh[4]  = (_Float16)(short)w1.x;  bh[5]  = (_Float16)(short)w1.y;
                bh[6]  = (_Float16)(short)w1.z;  bh[7]  = (_Float16)(short)w1.w;
                bh[8]  = (_Float16)(short)w2.x;  bh[9]  = (_Float16)(short)w2.y;
                bh[10] = (_Float16)(short)w2.z;  bh[11] = (_Float16)(short)w2.w;
                bh[12] = (_Float16)(short)w3.x;  bh[13] = (_Float16)(short)w3.y;
                bh[14] = (_Float16)(short)w3.z;  bh[15] = (_Float16)(short)w3.w;
                b[nj] = bh;
            }

            // ---- matrix core: 4 real WMMAs + 2 row-sum WMMAs ----
#pragma unroll
            for (int mi = 0; mi < 2; ++mi) {
                csum[mi] = __builtin_amdgcn_wmma_f32_16x16x32_f16(
                    false, a[mi], false, bones, (short)0, csum[mi], false, false);
#pragma unroll
                for (int nj = 0; nj < 2; ++nj) {
                    cg[mi][nj] = __builtin_amdgcn_wmma_f32_16x16x32_f16(
                        false, a[mi], false, b[nj], (short)0, cg[mi][nj], false, false);
                }
            }
        }

        // ---- fold group: acc += s * (x.wq) + z * rowsum(x) ----
        // C layout: n = lane&15 (both halves), m = vgpr r (+8 for hi half)
#pragma unroll
        for (int nj = 0; nj < 2; ++nj) {
            const int n = gn + nj * 16 + l16;
            const v2f sz = *(const v2f*)(saz + ((size_t)g * OUT_F + n) * 2);
            const float s = sz.x, z = sz.y;
#pragma unroll
            for (int mi = 0; mi < 2; ++mi) {
#pragma unroll
                for (int r = 0; r < 8; ++r) {
                    acc[mi][nj][r] += s * cg[mi][nj][r] + z * csum[mi][r];
                }
            }
        }
    }

    // ---- epilogue: bias add + store (f32 out) ----
#pragma unroll
    for (int nj = 0; nj < 2; ++nj) {
        const int n = gn + nj * 16 + l16;
        const float bb = bias[n];
#pragma unroll
        for (int mi = 0; mi < 2; ++mi) {
            const int mbase = gm + mi * 16 + (hiHalf ? 8 : 0);
#pragma unroll
            for (int r = 0; r < 8; ++r) {
                out[(size_t)(mbase + r) * OUT_F + n] = acc[mi][nj][r] + bb;
            }
        }
    }
}

extern "C" void kernel_launch(void* const* d_in, const int* in_sizes, int n_in,
                              void* d_out, int out_size, void* d_ws, size_t ws_size,
                              hipStream_t stream)
{
    (void)n_in; (void)out_size; (void)d_ws; (void)ws_size;
    const float* x    = (const float*)d_in[0];   // [4,2048,4096] f32
    const int*   w    = (const int*)  d_in[1];   // [11008,4096] int32 (int8 values)
    const float* saz  = (const float*)d_in[2];   // [32,11008,2] f32
    const float* bias = (const float*)d_in[3];   // [11008] f32
    float*       out  = (float*)d_out;           // [4,2048,11008] f32

    const int M = in_sizes[0] / IN_F;            // 8192
    dim3 grid(OUT_F / BN, M / BM);               // 86 x 128
    dim3 block(256);
    hipLaunchKernelGGL(w8g128_linear_wmma, grid, block, 0, stream,
                       x, w, saz, bias, out);
}
